// Sinkhorn_wZ_63891933496097
// MI455X (gfx1250) — compile-verified, matrix operand order checked by reading, not run on hardware
//
#include <hip/hip_runtime.h>

// ---------------------------------------------------------------------------
// Sizes (fixed by the reference): B=4, D=256, M=N=2048
// ---------------------------------------------------------------------------
constexpr int BATCH = 4;
constexpr int D     = 256;
constexpr int TOK   = 2048;
constexpr int MP    = 2049;          // TOK + 1 (dustbin row/col)
constexpr int LDP   = 2052;          // padded couplings row stride (16B-aligned rows)
constexpr int VSTR  = 2052;          // padded stride for u/v vectors (16B-aligned)
constexpr int NSEG  = 16;            // row segments for column-LSE pass
constexpr int SEGROWS = 129;         // ceil(2049/16)

constexpr float NORM      = -8.31776616671934371f;  // -log(M+N) = -log(4096)
constexpr float LOG_RATIO = -0.69314718055994531f;  // log(2048) - log(4096)
constexpr float SCALE     = 0.0625f;                // 1/sqrt(256)
constexpr float THR       = 0.2f;

typedef __attribute__((ext_vector_type(16))) __bf16 v16bf;
typedef __attribute__((ext_vector_type(8)))  __bf16 v8bf;
typedef __attribute__((ext_vector_type(8)))  float  v8f;

// ---------------------------------------------------------------------------
// Helpers: online log-sum-exp and block reductions (wave32, 8 waves/block)
// ---------------------------------------------------------------------------
__device__ inline void lse_push(float& m, float& s, float x) {
  if (x > m) { s = s * __expf(m - x) + 1.0f; m = x; }
  else       { s += __expf(x - m); }
}

__device__ inline void lse_combine(float& m, float& s, float m2, float s2) {
  float mx = fmaxf(m, m2);
  s = s * __expf(m - mx) + s2 * __expf(m2 - mx);
  m = mx;
}

// Result broadcast to all threads via shared memory (blockDim.x == 256).
__device__ inline float2 block_lse_reduce(float m, float s) {
  #pragma unroll
  for (int off = 16; off >= 1; off >>= 1) {
    float m2 = __shfl_xor(m, off, 32);
    float s2 = __shfl_xor(s, off, 32);
    lse_combine(m, s, m2, s2);
  }
  __shared__ float sm[8], ss[8];
  int wave = threadIdx.x >> 5, lane = threadIdx.x & 31;
  if (lane == 0) { sm[wave] = m; ss[wave] = s; }
  __syncthreads();
  if (threadIdx.x == 0) {
    m = sm[0]; s = ss[0];
    #pragma unroll
    for (int w = 1; w < 8; ++w) lse_combine(m, s, sm[w], ss[w]);
    sm[0] = m; ss[0] = s;
  }
  __syncthreads();
  return make_float2(sm[0], ss[0]);
}

// Valid in thread 0 after return.
__device__ inline void block_argmax(float& v, int& idx) {
  #pragma unroll
  for (int off = 16; off >= 1; off >>= 1) {
    float v2 = __shfl_xor(v, off, 32);
    int   i2 = __shfl_xor(idx, off, 32);
    if (v2 > v || (v2 == v && i2 < idx)) { v = v2; idx = i2; }
  }
  __shared__ float av[8];
  __shared__ int   ai[8];
  int wave = threadIdx.x >> 5, lane = threadIdx.x & 31;
  if (lane == 0) { av[wave] = v; ai[wave] = idx; }
  __syncthreads();
  if (threadIdx.x == 0) {
    #pragma unroll
    for (int w = 1; w < 8; ++w)
      if (av[w] > v || (av[w] == v && ai[w] < idx)) { v = av[w]; idx = ai[w]; }
  }
}

// Load a 16-bit WMMA operand fragment from K-contiguous (row/col-major) data.
// base points at (row-or-col r, k0 + 8*(lane>>4)).  Per the CDNA5 ISA layout,
// elements e=0..7 hold K = 8*hi + e and e=8..15 hold K = 16 + 8*hi + (e - 8).
__device__ inline v16bf load_frag_k16(const __bf16* base) {
  v8bf lo = *(const v8bf*)(base);
  v8bf hi = *(const v8bf*)(base + 16);
  v16bf f;
  #pragma unroll
  for (int e = 0; e < 8; ++e) { f[e] = lo[e]; f[e + 8] = hi[e]; }
  return f;
}

// ---------------------------------------------------------------------------
// Stage 0: conv_w f32 -> bf16
// ---------------------------------------------------------------------------
__global__ void cvt_w_kernel(const float* __restrict__ w, __bf16* __restrict__ wbf) {
  int i = blockIdx.x * 256 + threadIdx.x;   // grid covers exactly 256*256
  wbf[i] = (__bf16)w[i];
}

// ---------------------------------------------------------------------------
// Stage 1: mdescT[b, n, o] = sum_c conv_w[o,c] * desc[b,c,n] + conv_b[o]
// Output is token-major (D contiguous) bf16 so the scores GEMM fragments are
// contiguous b128 loads.  One wave per 16(token)x16(out-chan) tile, K=256.
// ---------------------------------------------------------------------------
__global__ void conv_kernel(const float* __restrict__ desc,
                            const __bf16* __restrict__ wbf,
                            const float* __restrict__ bias,
                            __bf16* __restrict__ outT) {
  int wave = threadIdx.x >> 5, lane = threadIdx.x & 31;
  int flat = blockIdx.x * 8 + wave;            // 4 * 128 * 16 = 8192 tiles
  int b    = flat >> 11;
  int rem  = flat & 2047;
  int tn   = (rem >> 4) << 4;                  // token tile base
  int to   = (rem & 15) << 4;                  // out-channel tile base
  int ll   = lane & 15;
  int hsel = lane >> 4;

  const float*  dbase = desc + (size_t)b * D * TOK + (tn + ll);   // + c*TOK
  const __bf16* wbase = wbf + (size_t)(to + ll) * D + hsel * 8;   // K contiguous

  v8f acc = {};
  for (int k0 = 0; k0 < D; k0 += 32) {
    v16bf af;
    #pragma unroll
    for (int e = 0; e < 8; ++e) {
      int c0 = k0 + hsel * 8 + e;
      int c1 = k0 + 16 + hsel * 8 + e;
      af[e]     = (__bf16)dbase[(size_t)c0 * TOK];   // strided gather (A^T)
      af[e + 8] = (__bf16)dbase[(size_t)c1 * TOK];
    }
    v16bf bfr = load_frag_k16(wbase + k0);
    acc = __builtin_amdgcn_wmma_f32_16x16x32_bf16(false, af, false, bfr,
                                                  (short)0, acc, false, false);
  }
  int   ocol = to + ll;
  float bv   = bias[ocol];
  #pragma unroll
  for (int r = 0; r < 8; ++r) {
    int tok = tn + r + (hsel << 3);
    outT[((size_t)b * TOK + tok) * D + ocol] = (__bf16)(acc[r] + bv);
  }
}

// ---------------------------------------------------------------------------
// Stage 2: per-(b,o) mean over tokens of mdesc0T  (coalesced across o)
// ---------------------------------------------------------------------------
__global__ void mean_kernel(const __bf16* __restrict__ outT, float* __restrict__ m0) {
  int b = blockIdx.x, o = threadIdx.x;        // 256 threads
  float s = 0.f;
  for (int n = 0; n < TOK; ++n)
    s += (float)outT[((size_t)b * TOK + n) * D + o];
  m0[b * D + o] = s * (1.0f / 2048.0f);
}

// ---------------------------------------------------------------------------
// Stage 3: tiny MLP -> z[b]   (m = concat([m0, m0]) per the reference)
// ---------------------------------------------------------------------------
__global__ void mlp_kernel(const float* __restrict__ m0,
                           const float* __restrict__ w1, const float* __restrict__ b1,
                           const float* __restrict__ w2, const float* __restrict__ b2,
                           const float* __restrict__ w3, const float* __restrict__ b3,
                           float* __restrict__ z) {
  int b = blockIdx.x, t = threadIdx.x;
  __shared__ float sm[256], h1[256], h2[64];
  sm[t] = m0[b * 256 + t];
  __syncthreads();
  float a1 = b1[t];
  for (int c = 0; c < 512; ++c) a1 += sm[c & 255] * w1[c * 256 + t];
  h1[t] = fmaxf(a1, 0.f);
  __syncthreads();
  if (t < 64) {
    float a2 = b2[t];
    for (int c = 0; c < 256; ++c) a2 += h1[c] * w2[c * 64 + t];
    h2[t] = fmaxf(a2, 0.f);
  }
  __syncthreads();
  if (t == 0) {
    float a3 = b3[0];
    for (int c = 0; c < 64; ++c) a3 += h2[c] * w3[c];
    z[b] = a3;
  }
}

// ---------------------------------------------------------------------------
// Stage 4: scores GEMM -> inner couplings (f32), row stride ldc
// scores[b, n, m] = (1/16) * sum_d mdesc0T[b,n,d] * mdesc1T[b,m,d]
// One wave per 16x16 tile; both fragments are two b128 loads per K-step.
// ---------------------------------------------------------------------------
__global__ void scores_kernel(const __bf16* __restrict__ A0T,
                              const __bf16* __restrict__ A1T,
                              float* __restrict__ C, int ldc) {
  int wave = threadIdx.x >> 5, lane = threadIdx.x & 31;
  int flat = blockIdx.x * 8 + wave;            // 4 * 128 * 128 = 65536 tiles
  int b    = flat >> 14;
  int rem  = flat & 16383;
  int tn   = (rem >> 7) << 4;                  // desc0-token (row) tile
  int tm   = (rem & 127) << 4;                 // desc1-token (col) tile
  int ll   = lane & 15;
  int hb   = (lane >> 4) * 8;

  const __bf16* a0 = A0T + ((size_t)b * TOK + tn + ll) * D + hb;
  const __bf16* b0 = A1T + ((size_t)b * TOK + tm + ll) * D + hb;

  v8f acc = {};
  #pragma unroll
  for (int k0 = 0; k0 < D; k0 += 32) {
    v16bf af  = load_frag_k16(a0 + k0);
    v16bf bfr = load_frag_k16(b0 + k0);
    acc = __builtin_amdgcn_wmma_f32_16x16x32_bf16(false, af, false, bfr,
                                                  (short)0, acc, false, false);
  }
  float* crow = C + (size_t)b * MP * ldc + (size_t)tn * ldc + tm;
  #pragma unroll
  for (int r = 0; r < 8; ++r) {
    int row = r + ((lane >> 4) << 3);
    crow[(size_t)row * ldc + ll] = acc[r] * SCALE;
  }
}

// ---------------------------------------------------------------------------
// Stage 5: dustbin row/col = z[b]
// ---------------------------------------------------------------------------
__global__ void border_kernel(const float* __restrict__ z, float* __restrict__ C,
                              int ldc) {
  int b = blockIdx.y;
  int j = blockIdx.x * 256 + threadIdx.x;
  if (j >= MP) return;
  float  zv   = z[b];
  float* base = C + (size_t)b * MP * ldc;
  base[(size_t)(MP - 1) * ldc + j] = zv;
  base[(size_t)j * ldc + (MP - 1)] = zv;
}

__global__ void zero_kernel(float* __restrict__ p, int n) {
  int i = blockIdx.x * 256 + threadIdx.x;
  if (i < n) p[i] = 0.f;
}

// ---------------------------------------------------------------------------
// Sinkhorn sweeps.  Couplings stay resident in L2 (67MB < 192MB global L2);
// the limiter is instruction issue, so the padded-stride variants use b128
// (float4) loads -- rows of the padded buffer are 16B-aligned (ldp = 2052).
// u[b,i] = log_mu[i] - LSE_j( C[b,i,j] + v[b,j] )
// ---------------------------------------------------------------------------
__global__ void u_vec4_kernel(const float* __restrict__ C, int ldp,
                              const float* __restrict__ v, float* __restrict__ u) {
  int i = blockIdx.x, b = blockIdx.y, t = threadIdx.x;
  const float* row = C + (size_t)b * MP * ldp + (size_t)i * ldp;
  const float* vb  = v + b * VSTR;
  float m = -__builtin_inff(), s = 0.f;
  #pragma unroll
  for (int g = t; g < 512; g += 256) {        // exactly 2 iterations per thread
    float4 x  = *(const float4*)(row + 4 * g);
    float4 vv = *(const float4*)(vb + 4 * g);
    lse_push(m, s, x.x + vv.x);
    lse_push(m, s, x.y + vv.y);
    lse_push(m, s, x.z + vv.z);
    lse_push(m, s, x.w + vv.w);
  }
  if (t == 0) lse_push(m, s, row[2048] + vb[2048]);   // tail column
  float2 r = block_lse_reduce(m, s);
  if (t == 0) {
    float logmu = (i < TOK) ? NORM : LOG_RATIO;
    u[b * VSTR + i] = logmu - (r.x + __logf(r.y));
  }
}

// Column LSE over a row segment; 4 columns per thread (independent online
// accumulators), one b128 load per row per thread.  Column group 512 covers
// the dustbin column 2048 plus 3 padding columns whose (finite) results are
// discarded by the j<MP guard in v_comb.
__global__ void v_part_vec4_kernel(const float* __restrict__ C, int ldp,
                                   const float* __restrict__ u,
                                   float* __restrict__ pm, float* __restrict__ ps) {
  int g = blockIdx.x * 256 + threadIdx.x;     // column group: 0..512
  int seg = blockIdx.y, b = blockIdx.z;
  if (g > 512) return;
  int i0 = seg * SEGROWS;
  int i1 = min(i0 + SEGROWS, MP);
  const float* base = C + (size_t)b * MP * ldp + 4 * g;
  const float* ub   = u + b * VSTR;
  float m0 = -__builtin_inff(), s0 = 0.f, m1 = -__builtin_inff(), s1 = 0.f;
  float m2 = -__builtin_inff(), s2 = 0.f, m3 = -__builtin_inff(), s3 = 0.f;
  for (int i = i0; i < i1; ++i) {
    float4 x = *(const float4*)(base + (size_t)i * ldp);
    float uu = ub[i];
    lse_push(m0, s0, x.x + uu);
    lse_push(m1, s1, x.y + uu);
    lse_push(m2, s2, x.z + uu);
    lse_push(m3, s3, x.w + uu);
  }
  size_t o = (size_t)(b * NSEG + seg) * VSTR + 4 * g;   // 16B aligned
  *(float4*)(pm + o) = make_float4(m0, m1, m2, m3);
  *(float4*)(ps + o) = make_float4(s0, s1, s2, s3);
}

// Scalar fallbacks (used when ws is too small for the padded couplings copy).
__global__ void u_kernel(const float* __restrict__ C, int ldc,
                         const float* __restrict__ v, float* __restrict__ u) {
  int i = blockIdx.x, b = blockIdx.y, t = threadIdx.x;
  const float* row = C + (size_t)b * MP * ldc + (size_t)i * ldc;
  const float* vb  = v + b * VSTR;
  float m = -__builtin_inff(), s = 0.f;
  for (int j = t; j < MP; j += 256) lse_push(m, s, row[j] + vb[j]);
  float2 r = block_lse_reduce(m, s);
  if (t == 0) {
    float logmu = (i < TOK) ? NORM : LOG_RATIO;
    u[b * VSTR + i] = logmu - (r.x + __logf(r.y));
  }
}

__global__ void v_part_kernel(const float* __restrict__ C, int ldc,
                              const float* __restrict__ u,
                              float* __restrict__ pm, float* __restrict__ ps) {
  int j = blockIdx.x * 256 + threadIdx.x;
  int seg = blockIdx.y, b = blockIdx.z;
  if (j >= MP) return;
  int i0 = seg * SEGROWS;
  int i1 = min(i0 + SEGROWS, MP);
  const float* base = C + (size_t)b * MP * ldc;
  const float* ub   = u + b * VSTR;
  float m = -__builtin_inff(), s = 0.f;
  for (int i = i0; i < i1; ++i) lse_push(m, s, base[(size_t)i * ldc + j] + ub[i]);
  size_t o = (size_t)(b * NSEG + seg) * VSTR + j;
  pm[o] = m; ps[o] = s;
}

__global__ void v_comb_kernel(const float* __restrict__ pm, const float* __restrict__ ps,
                              float* __restrict__ v) {
  int j = blockIdx.x * 256 + threadIdx.x;
  int b = blockIdx.y;
  if (j >= MP) return;
  size_t base = (size_t)b * NSEG * VSTR + j;
  float m = pm[base], s = ps[base];
  #pragma unroll
  for (int seg = 1; seg < NSEG; ++seg)
    lse_combine(m, s, pm[base + (size_t)seg * VSTR], ps[base + (size_t)seg * VSTR]);
  float lognu = (j < TOK) ? NORM : LOG_RATIO;
  v[b * VSTR + j] = lognu - (m + __logf(s));
}

// ---------------------------------------------------------------------------
// Stage 6: row softmax from couplings (stride ldc) into S (stride MP).
// u_i and -norm cancel inside a row; only v_j matters.  Also records the
// inner (j<2048) row max / argmax.  Works in-place when Csrc == Sdst.
// ---------------------------------------------------------------------------
__global__ void softmax_kernel(const float* __restrict__ v,
                               const float* __restrict__ Csrc, int ldc,
                               float* __restrict__ Sdst,
                               float* __restrict__ smax, int* __restrict__ sidx) {
  int i = blockIdx.x, b = blockIdx.y, t = threadIdx.x;
  const float* row = Csrc + (size_t)b * MP * ldc + (size_t)i * ldc;
  float*       dst = Sdst + (size_t)b * MP * MP + (size_t)i * MP;
  const float* vb  = v + b * VSTR;
  float m = -__builtin_inff(), s = 0.f;
  float bmax = -__builtin_inff(); int bidx = 0;
  for (int j = t; j < MP; j += 256) {
    float x = row[j] + vb[j];
    lse_push(m, s, x);
    if (j < TOK && x > bmax) { bmax = x; bidx = j; }
  }
  float2 r = block_lse_reduce(m, s);
  float L = r.x + __logf(r.y);
  for (int j = t; j < MP; j += 256)
    dst[j] = __expf(row[j] + vb[j] - L);
  block_argmax(bmax, bidx);
  if (t == 0 && i < TOK) {
    smax[b * TOK + i] = __expf(bmax - L);   // S max value in the row
    sidx[b * TOK + i] = bidx;
  }
}

__global__ void colmax_kernel(const float* __restrict__ C, int* __restrict__ idx1) {
  int j = blockIdx.x * 256 + threadIdx.x;
  int b = blockIdx.y;
  if (j >= TOK) return;
  const float* base = C + (size_t)b * MP * MP + j;
  float best = -__builtin_inff(); int bi = 0;
  for (int i = 0; i < TOK; ++i) {
    float x = base[(size_t)i * MP];
    if (x > best) { best = x; bi = i; }
  }
  idx1[b * TOK + j] = bi;
}

// ---------------------------------------------------------------------------
// Stage 7: mutual matching
// ---------------------------------------------------------------------------
__global__ void match0_kernel(const float* __restrict__ smax, const int* __restrict__ idx0,
                              const int* __restrict__ idx1,
                              float* __restrict__ matches0, float* __restrict__ ms0) {
  int i = blockIdx.x * 256 + threadIdx.x;
  int b = blockIdx.y;
  if (i >= TOK) return;
  int  k      = idx0[b * TOK + i];
  bool mutual = (idx1[b * TOK + k] == i);
  float sc    = mutual ? __expf(smax[b * TOK + i]) : 0.f;  // exp(max0), per ref
  ms0[b * TOK + i]      = sc;
  matches0[b * TOK + i] = (mutual && sc > THR) ? (float)k : -1.f;
}

__global__ void match1_kernel(const int* __restrict__ idx0, const int* __restrict__ idx1,
                              const float* __restrict__ ms0,
                              float* __restrict__ matches1, float* __restrict__ ms1) {
  int j = blockIdx.x * 256 + threadIdx.x;
  int b = blockIdx.y;
  if (j >= TOK) return;
  int  k       = idx1[b * TOK + j];
  bool mutual1 = (idx0[b * TOK + k] == j);
  float sc0    = ms0[b * TOK + k];
  ms1[b * TOK + j]      = mutual1 ? sc0 : 0.f;
  matches1[b * TOK + j] = (mutual1 && sc0 > THR) ? (float)k : -1.f;
}

// ---------------------------------------------------------------------------
// Host orchestration
// ---------------------------------------------------------------------------
extern "C" void kernel_launch(void* const* d_in, const int* in_sizes, int n_in,
                              void* d_out, int out_size, void* d_ws, size_t ws_size,
                              hipStream_t stream) {
  (void)in_sizes; (void)n_in; (void)out_size;

  const float* desc0 = (const float*)d_in[0];
  const float* desc1 = (const float*)d_in[1];
  const float* convw = (const float*)d_in[2];
  const float* convb = (const float*)d_in[3];
  const float* zw1 = (const float*)d_in[4];
  const float* zb1 = (const float*)d_in[5];
  const float* zw2 = (const float*)d_in[6];
  const float* zb2 = (const float*)d_in[7];
  const float* zw3 = (const float*)d_in[8];
  const float* zb3 = (const float*)d_in[9];

  float* out      = (float*)d_out;
  float* matches0 = out;
  float* matches1 = out + BATCH * TOK;
  float* ms0      = out + 2 * BATCH * TOK;
  float* ms1      = out + 3 * BATCH * TOK;
  float* S        = out + 4 * BATCH * TOK;   // (B, 2049, 2049) final softmax

  // Workspace layout (bytes)
  char* ws = (char*)d_ws;
  constexpr size_t OFF_WBF = 0;                                  // 256*256 bf16
  constexpr size_t OFF_A0  = OFF_WBF + (size_t)D * D * 2;        // B*TOK*D bf16
  constexpr size_t OFF_A1  = OFF_A0 + (size_t)BATCH * TOK * D * 2;
  constexpr size_t OFF_U   = OFF_A1 + (size_t)BATCH * TOK * D * 2;
  constexpr size_t OFF_V   = OFF_U + (size_t)BATCH * VSTR * 4;   // u,v contiguous
  constexpr size_t OFF_PM  = OFF_V + (size_t)BATCH * VSTR * 4;
  constexpr size_t OFF_PS  = OFF_PM + (size_t)BATCH * NSEG * VSTR * 4;
  constexpr size_t OFF_M0  = OFF_PS + (size_t)BATCH * NSEG * VSTR * 4;
  constexpr size_t OFF_Z   = OFF_M0 + (size_t)BATCH * D * 4;
  constexpr size_t OFF_SMX = OFF_Z + 256;                        // padded
  constexpr size_t OFF_I0  = OFF_SMX + (size_t)BATCH * TOK * 4;
  constexpr size_t OFF_I1  = OFF_I0 + (size_t)BATCH * TOK * 4;
  constexpr size_t OFF_CP  = (OFF_I1 + (size_t)BATCH * TOK * 4 + 255) & ~(size_t)255;
  constexpr size_t WS_NEED_PADDED = OFF_CP + (size_t)BATCH * MP * LDP * 4;

  __bf16* wbf = (__bf16*)(ws + OFF_WBF);
  __bf16* A0T = (__bf16*)(ws + OFF_A0);
  __bf16* A1T = (__bf16*)(ws + OFF_A1);
  float*  u   = (float*)(ws + OFF_U);
  float*  v   = (float*)(ws + OFF_V);
  float*  pm  = (float*)(ws + OFF_PM);
  float*  ps  = (float*)(ws + OFF_PS);
  float*  m0  = (float*)(ws + OFF_M0);
  float*  z   = (float*)(ws + OFF_Z);
  float*  smx = (float*)(ws + OFF_SMX);
  int*    i0  = (int*)(ws + OFF_I0);
  int*    i1  = (int*)(ws + OFF_I1);

  // Padded couplings copy (stride 2052 -> 16B-aligned rows) enables b128
  // streaming in the Sinkhorn sweeps; fall back to in-place (stride 2049,
  // scalar b32) if the workspace is too small.  ws_size is fixed per run,
  // so this choice is deterministic.
  const bool padded = (ws_size >= WS_NEED_PADDED);
  float* CP  = padded ? (float*)(ws + OFF_CP) : S;
  const int ldc = padded ? LDP : MP;

  // Stage 0-3: weights->bf16, conv GEMMs (WMMA), means, MLP -> z
  cvt_w_kernel<<<dim3(D * D / 256), dim3(256), 0, stream>>>(convw, wbf);
  conv_kernel<<<dim3(1024), dim3(256), 0, stream>>>(desc0, wbf, convb, A0T);
  conv_kernel<<<dim3(1024), dim3(256), 0, stream>>>(desc1, wbf, convb, A1T);
  mean_kernel<<<dim3(BATCH), dim3(256), 0, stream>>>(A0T, m0);
  mlp_kernel<<<dim3(BATCH), dim3(256), 0, stream>>>(m0, zw1, zb1, zw2, zb2, zw3, zb3, z);

  // Stage 4-5: couplings (scores via WMMA + z borders), u=v=0
  scores_kernel<<<dim3(8192), dim3(256), 0, stream>>>(A0T, A1T, CP, ldc);
  border_kernel<<<dim3((MP + 255) / 256, BATCH), dim3(256), 0, stream>>>(z, CP, ldc);
  {
    int n = 2 * BATCH * VSTR;   // u and v are contiguous
    zero_kernel<<<dim3((n + 255) / 256), dim3(256), 0, stream>>>(u, n);
  }

  // Sinkhorn: 100 iterations, streaming from L2
  for (int it = 0; it < 100; ++it) {
    if (padded) {
      u_vec4_kernel<<<dim3(MP, BATCH), dim3(256), 0, stream>>>(CP, ldc, v, u);
      v_part_vec4_kernel<<<dim3(3, NSEG, BATCH), dim3(256), 0, stream>>>(CP, ldc, u, pm, ps);
    } else {
      u_kernel<<<dim3(MP, BATCH), dim3(256), 0, stream>>>(CP, ldc, v, u);
      v_part_kernel<<<dim3((MP + 255) / 256, NSEG, BATCH), dim3(256), 0, stream>>>(CP, ldc, u, pm, ps);
    }
    v_comb_kernel<<<dim3((MP + 255) / 256, BATCH), dim3(256), 0, stream>>>(pm, ps, v);
  }

  // Stage 6-7: softmax into S + matching
  softmax_kernel<<<dim3(MP, BATCH), dim3(256), 0, stream>>>(v, CP, ldc, S, smx, i0);
  colmax_kernel<<<dim3(TOK / 256, BATCH), dim3(256), 0, stream>>>(S, i1);
  match0_kernel<<<dim3(TOK / 256, BATCH), dim3(256), 0, stream>>>(smx, i0, i1, matches0, ms0);
  match1_kernel<<<dim3(TOK / 256, BATCH), dim3(256), 0, stream>>>(i0, i1, ms0, matches1, ms1);
}
